// DGCNN_59399397704323
// MI455X (gfx1250) — matrix operand order repaired and checked by reading, not compiled, not run on hardware
//
#include <hip/hip_runtime.h>
#include <hip/hip_bf16.h>

typedef float v8f  __attribute__((ext_vector_type(8)));
typedef __bf16 v16bf __attribute__((ext_vector_type(16)));
typedef __bf16 v8bf  __attribute__((ext_vector_type(8)));

#define BATCH 4
#define NPTS  1024
#define KNN   20
#define NEG_INF (-3.0e38f)

__device__ __forceinline__ v16bf cat8(v8bf lo, v8bf hi) {
    v16bf r;
#pragma unroll
    for (int i = 0; i < 8; ++i) { r[i] = lo[i]; r[8 + i] = hi[i]; }
    return r;
}

__device__ __forceinline__ v8f wmma_bf16(v16bf a, v16bf b, v8f c) {
    return __builtin_amdgcn_wmma_f32_16x16x32_bf16(false, a, false, b, (short)0, c, false, false);
}

__device__ __forceinline__ float leaky(float z) { return z >= 0.0f ? z : 0.2f * z; }

// ---------------------------------------------------------------------------
// Pack x (B,N,3) f32 -> F0 (B*N, 32) bf16, channels 3..31 zero.
// ---------------------------------------------------------------------------
__global__ void pack_x_kernel(const float* __restrict__ x, __bf16* __restrict__ F0) {
    int p = blockIdx.x * blockDim.x + threadIdx.x;
    if (p >= BATCH * NPTS) return;
#pragma unroll
    for (int c = 0; c < 32; ++c)
        F0[(size_t)p * 32 + c] = (__bf16)(c < 3 ? x[(size_t)p * 3 + c] : 0.0f);
}

// ---------------------------------------------------------------------------
// Repack conv weight (O, 2*Cin) f32 -> (O, 2*CS) bf16 with zero padding.
// Packed K order: k<CS -> edge half (W1), k>=CS -> center half (W2).
// ---------------------------------------------------------------------------
__global__ void repack_w_kernel(const float* __restrict__ W, __bf16* __restrict__ Wb,
                                int O, int Cin, int CS) {
    int t = blockIdx.x * blockDim.x + threadIdx.x;
    int tot = O * 2 * CS;
    if (t >= tot) return;
    int o = t / (2 * CS);
    int k = t - o * 2 * CS;
    float v;
    if (k < CS) v = (k < Cin) ? W[(size_t)o * 2 * Cin + k] : 0.0f;
    else {
        int k2 = k - CS;
        v = (k2 < Cin) ? W[(size_t)o * 2 * Cin + Cin + k2] : 0.0f;
    }
    Wb[t] = (__bf16)v;
}

__global__ void cvt_bf16_kernel(const float* __restrict__ s, __bf16* __restrict__ d, int n) {
    int t = blockIdx.x * blockDim.x + threadIdx.x;
    if (t < n) d[t] = (__bf16)s[t];
}

// ---------------------------------------------------------------------------
// Per-point squared norm of bf16 features (f32 accumulate), vectorized.
// ---------------------------------------------------------------------------
__global__ void xx_kernel(const __bf16* __restrict__ F, int fstride, int C,
                          float* __restrict__ xx) {
    int p = blockIdx.x * blockDim.x + threadIdx.x;
    if (p >= BATCH * NPTS) return;
    const __bf16* row = F + (size_t)p * fstride;
    float s = 0.0f;
    for (int k8 = 0; k8 < C; k8 += 8) {
        v8bf v = *(const v8bf*)(row + k8);
#pragma unroll
        for (int i = 0; i < 8; ++i) { float f = (float)v[i]; s += f * f; }
    }
    xx[p] = s;
}

// ---------------------------------------------------------------------------
// Gram + top-K: one block = 16 query rows of one batch, 512 threads = 16 waves.
// Wave w owns m-tiles {w, w+16, w+32, w+48}; A tile hoisted, 4 accumulators.
// pd[16][1024] f32 in LDS; then wave w does 20 rounds of argmax for row w.
// ---------------------------------------------------------------------------
__global__ void __launch_bounds__(512)
gram_knn_kernel(const __bf16* __restrict__ F, int fstride, int C,
                const float* __restrict__ xx, int* __restrict__ idxOut) {
    __shared__ float pd[16 * 1024];

    int bid = blockIdx.x;
    int b = bid >> 6;            // N/16 = 64 row-tiles per batch
    int nt = bid & 63;
    int n0 = nt * 16;
    const __bf16* Fb = F + (size_t)b * NPTS * fstride;
    const float* xb = xx + b * NPTS;

    int tid = threadIdx.x;
    int wv = tid >> 5, lane = tid & 31, q = lane & 15, hk = lane >> 4;

    v8f acc[4] = {};
    for (int kt = 0; kt < (C >> 5); ++kt) {
        int k0 = kt * 32;
        // A tile (shared across the wave's 4 m-tiles): rows n0..n0+15
        const __bf16* arow = Fb + (size_t)(n0 + q) * fstride + k0;
        v8bf alo = *(const v8bf*)(arow + hk * 16);
        v8bf ahi = *(const v8bf*)(arow + 8 + hk * 16);
        v16bf a = cat8(alo, ahi);
#pragma unroll
        for (int t = 0; t < 4; ++t) {
            int m0 = (wv + 16 * t) * 16;
            v16bf bm = *(const v16bf*)(Fb + (size_t)(m0 + q) * fstride + k0 + hk * 16);
            acc[t] = wmma_bf16(a, bm, acc[t]);
        }
    }
#pragma unroll
    for (int t = 0; t < 4; ++t) {
        int m0 = (wv + 16 * t) * 16;
        float xm = xb[m0 + q];
#pragma unroll
        for (int r = 0; r < 8; ++r) {
            int row = r + 8 * hk;
            pd[row * 1024 + (m0 + q)] = 2.0f * acc[t][r] - xb[n0 + row] - xm;
        }
    }
    __syncthreads();

    // top-20 for row `wv` (one wave per row), smallest-index tie-break
    float* prow = pd + wv * 1024;
    int* orow = idxOut + ((size_t)(b * NPTS + n0 + wv)) * KNN;
    for (int t = 0; t < KNN; ++t) {
        float best = NEG_INF;
        int bi = 1 << 30;
        for (int i = 0; i < 32; ++i) {
            int c = i * 32 + lane;
            float v = prow[c];
            if (v > best || (v == best && c < bi)) { best = v; bi = c; }
        }
#pragma unroll
        for (int off = 16; off; off >>= 1) {
            float ov = __shfl_xor(best, off);
            int oi = __shfl_xor(bi, off);
            if (ov > best || (ov == best && oi < bi)) { best = ov; bi = oi; }
        }
        if (lane == 0) orow[t] = bi;
        if ((bi & 31) == lane) prow[bi] = NEG_INF;   // owner retires the winner
        __syncthreads();                             // uniform across all 16 waves
    }
}

// ---------------------------------------------------------------------------
// Y2 GEMM: y2[p][o] = W2[o][:] . center_feat[p][:]  (dense over all points).
// One block = 16 points, 8 waves, MT = O/128 M-tiles per wave.
// ---------------------------------------------------------------------------
template <int CS, int O>
__global__ void __launch_bounds__(256)
y2_gemm_kernel(const __bf16* __restrict__ F, int fstride,
               const __bf16* __restrict__ Wb,   // packed [O][2*CS]; W2 at col CS
               float* __restrict__ Y2) {        // [B*N][O]
    int p0 = blockIdx.x * 16;
    int tid = threadIdx.x;
    int wv = tid >> 5, lane = tid & 31, q = lane & 15, hk = lane >> 4;
    constexpr int MT = O / 128;

    v8f acc[MT] = {};
    for (int kt = 0; kt < CS / 32; ++kt) {
        int k0 = kt * 32;
        v16bf bm = *(const v16bf*)(F + (size_t)(p0 + q) * fstride + k0 + hk * 16);
#pragma unroll
        for (int t = 0; t < MT; ++t) {
            int mt = wv + 8 * t;
            const __bf16* wrow = Wb + (size_t)(mt * 16 + q) * (2 * CS) + CS;
            v8bf alo = *(const v8bf*)(wrow + k0 + hk * 16);
            v8bf ahi = *(const v8bf*)(wrow + k0 + 8 + hk * 16);
            acc[t] = wmma_bf16(cat8(alo, ahi), bm, acc[t]);
        }
    }
#pragma unroll
    for (int t = 0; t < MT; ++t) {
        int mt = wv + 8 * t;
#pragma unroll
        for (int r = 0; r < 8; ++r) {
            int o = mt * 16 + r + 8 * hk;
            Y2[(size_t)(p0 + q) * O + o] = acc[t][r];
        }
    }
}

// ---------------------------------------------------------------------------
// EdgeConv (edge half only): one block per point. LDS E1 = 20 rows x CS bf16
// (nb_j - center). K loop over CS only; y2 (= W2*center) read from Y2 buffer.
// B tiles hoisted out of the M-tile loop; register-blocked accumulators.
// ---------------------------------------------------------------------------
template <int CS, int O>
__global__ void __launch_bounds__(256)
edgeconv_kernel(const __bf16* __restrict__ F, int fstride,
                const __bf16* __restrict__ Wb,
                const float* __restrict__ g, const float* __restrict__ beta,
                const int* __restrict__ idx,
                const float* __restrict__ Y2,
                __bf16* __restrict__ out /* = Fcat + channel offset */) {
    __shared__ __align__(32) __bf16 E1[20 * CS];
    __shared__ int nidx[KNN];

    int p = blockIdx.x;
    int b = p >> 10, n = p & 1023;
    const __bf16* Fb = F + (size_t)b * NPTS * fstride;
    int tid = threadIdx.x;

    if (tid < KNN) nidx[tid] = idx[(size_t)p * KNN + tid];
    __syncthreads();

    // fill E1 with 16B vector ops
    const __bf16* crow = Fb + (size_t)n * fstride;
    constexpr int C8 = CS / 8;
    for (int it = tid; it < KNN * C8; it += 256) {
        int j = it / C8;
        int k8 = (it - j * C8) * 8;
        const __bf16* nrow = Fb + (size_t)nidx[j] * fstride;
        v8bf nb = *(const v8bf*)(nrow + k8);
        v8bf cn = *(const v8bf*)(crow + k8);
        v8bf d;
#pragma unroll
        for (int i = 0; i < 8; ++i) d[i] = (__bf16)((float)nb[i] - (float)cn[i]);
        *(v8bf*)&E1[j * CS + k8] = d;
    }
    __syncthreads();

    int wv = tid >> 5, lane = tid & 31, q = lane & 15, hk = lane >> 4;
    __bf16* orow = out + (size_t)p * 1920;
    const float* y2row = Y2 + (size_t)p * O;
    v16bf zero = {};
    constexpr int MT = O / 128;

    v8f acc0[MT] = {}, acc1[MT] = {};
    for (int kt = 0; kt < CS / 32; ++kt) {
        int k0 = kt * 32;
        int kv = k0 + hk * 16;
        v16bf b0 = *(const v16bf*)(&E1[q * CS + kv]);                       // edges 0..15
        v16bf b1 = (q < 4) ? *(const v16bf*)(&E1[(16 + q) * CS + kv]) : zero; // 16..19
#pragma unroll
        for (int t = 0; t < MT; ++t) {
            int mt = wv + 8 * t;
            const __bf16* wrow = Wb + (size_t)(mt * 16 + q) * (2 * CS);
            v8bf alo = *(const v8bf*)(wrow + k0 + hk * 16);
            v8bf ahi = *(const v8bf*)(wrow + k0 + 8 + hk * 16);
            v16bf a = cat8(alo, ahi);
            acc0[t] = wmma_bf16(a, b0, acc0[t]);
            acc1[t] = wmma_bf16(a, b1, acc1[t]);
        }
    }
#pragma unroll
    for (int t = 0; t < MT; ++t) {
        int mt = wv + 8 * t;
#pragma unroll
        for (int r = 0; r < 8; ++r) {
            int o = mt * 16 + r + 8 * hk;
            float y2 = y2row[o];
            float gg = g[o], bb = beta[o];
            float z0 = leaky(gg * (acc0[t][r] + y2) + bb);   // edges 0..15
            float z1 = leaky(gg * (acc1[t][r] + y2) + bb);   // edges 16..19
            if (q >= 4) z1 = NEG_INF;
            float z = fmaxf(z0, z1);
#pragma unroll
            for (int off = 8; off; off >>= 1) z = fmaxf(z, __shfl_xor(z, off));
            if (q == 0) orow[o] = (__bf16)z;
        }
    }
}

// ---------------------------------------------------------------------------
// Final 1x1 conv: out[b][n][o] = leaky(gl*(Wl x cat) + bl), f32 output.
// One block = 16 points; 8 waves x 4 M-tiles each; B tile hoisted.
// ---------------------------------------------------------------------------
__global__ void __launch_bounds__(256)
final_gemm_kernel(const __bf16* __restrict__ Fcat, const __bf16* __restrict__ Wlb,
                  const float* __restrict__ gl, const float* __restrict__ bl,
                  float* __restrict__ outp) {
    int p0 = blockIdx.x * 16;
    int tid = threadIdx.x;
    int wv = tid >> 5, lane = tid & 31, q = lane & 15, hk = lane >> 4;
    const __bf16* Fb = Fcat + (size_t)p0 * 1920;

    v8f acc[4] = {};
    for (int kt = 0; kt < 60; ++kt) {
        int k0 = kt * 32;
        v16bf bm = *(const v16bf*)(Fb + (size_t)q * 1920 + k0 + hk * 16);
#pragma unroll
        for (int t = 0; t < 4; ++t) {
            int mt = wv + 8 * t;
            const __bf16* wrow = Wlb + (size_t)(mt * 16 + q) * 1920;
            v8bf alo = *(const v8bf*)(wrow + k0 + hk * 16);
            v8bf ahi = *(const v8bf*)(wrow + k0 + 8 + hk * 16);
            acc[t] = wmma_bf16(cat8(alo, ahi), bm, acc[t]);
        }
    }
#pragma unroll
    for (int t = 0; t < 4; ++t) {
        int mt = wv + 8 * t;
#pragma unroll
        for (int r = 0; r < 8; ++r) {
            int o = mt * 16 + r + 8 * hk;
            float z = leaky(gl[o] * acc[t][r] + bl[o]);
            outp[(size_t)(p0 + q) * 512 + o] = z;
        }
    }
}

// ---------------------------------------------------------------------------
extern "C" void kernel_launch(void* const* d_in, const int* in_sizes, int n_in,
                              void* d_out, int out_size, void* d_ws, size_t ws_size,
                              hipStream_t stream) {
    (void)in_sizes; (void)n_in; (void)out_size; (void)ws_size;

    const float* x  = (const float*)d_in[0];
    const float* W[4]  = {(const float*)d_in[1], (const float*)d_in[4],
                          (const float*)d_in[7], (const float*)d_in[10]};
    const float* g[4]  = {(const float*)d_in[2], (const float*)d_in[5],
                          (const float*)d_in[8], (const float*)d_in[11]};
    const float* bt[4] = {(const float*)d_in[3], (const float*)d_in[6],
                          (const float*)d_in[9], (const float*)d_in[12]};
    const float* Wl = (const float*)d_in[13];
    const float* gl = (const float*)d_in[14];
    const float* bl = (const float*)d_in[15];
    float* out = (float*)d_out;

    const int CS[4]  = {32, 128, 256, 512};    // stored input channels per layer
    const int Cin[4] = {3, 128, 256, 512};
    const int O[4]   = {128, 256, 512, 1024};
    const int off[4] = {0, 128, 384, 896};     // channel offsets in the 1920 cat

    // workspace carve-up (all 256B aligned)
    char* ws = (char*)d_ws;
    size_t cur = 0;
    auto carve = [&](size_t bytes) {
        char* p = ws + cur;
        cur += (bytes + 255) & ~(size_t)255;
        return p;
    };
    __bf16* F0   = (__bf16*)carve((size_t)BATCH * NPTS * 32 * 2);
    __bf16* Fcat = (__bf16*)carve((size_t)BATCH * NPTS * 1920 * 2);
    __bf16* Wb[4];
    for (int l = 0; l < 4; ++l) Wb[l] = (__bf16*)carve((size_t)O[l] * 2 * CS[l] * 2);
    __bf16* Wlb  = (__bf16*)carve((size_t)512 * 1920 * 2);
    float*  xxp  = (float*)carve((size_t)BATCH * NPTS * 4);
    int*    idxp = (int*)carve((size_t)BATCH * NPTS * KNN * 4);
    float*  y2p  = (float*)carve((size_t)BATCH * NPTS * 1024 * 4);  // max O

    // --- weight / input preprocessing (bf16 repack) ---
    pack_x_kernel<<<(BATCH * NPTS + 255) / 256, 256, 0, stream>>>(x, F0);
    for (int l = 0; l < 4; ++l) {
        int tot = O[l] * 2 * CS[l];
        repack_w_kernel<<<(tot + 255) / 256, 256, 0, stream>>>(W[l], Wb[l], O[l], Cin[l], CS[l]);
    }
    cvt_bf16_kernel<<<(512 * 1920 + 255) / 256, 256, 0, stream>>>(Wl, Wlb, 512 * 1920);

    // --- 4 EdgeConv layers ---
    for (int l = 0; l < 4; ++l) {
        const __bf16* Fin = (l == 0) ? F0 : (Fcat + off[l - 1]);
        int stride = (l == 0) ? 32 : 1920;

        xx_kernel<<<(BATCH * NPTS + 255) / 256, 256, 0, stream>>>(Fin, stride, CS[l], xxp);
        gram_knn_kernel<<<BATCH * (NPTS / 16), 512, 0, stream>>>(Fin, stride, CS[l], xxp, idxp);

        __bf16* outSlice = Fcat + off[l];
        switch (l) {
            case 0:
                y2_gemm_kernel<32, 128><<<BATCH * NPTS / 16, 256, 0, stream>>>(Fin, stride, Wb[0], y2p);
                edgeconv_kernel<32, 128><<<BATCH * NPTS, 256, 0, stream>>>(
                    Fin, stride, Wb[0], g[0], bt[0], idxp, y2p, outSlice);
                break;
            case 1:
                y2_gemm_kernel<128, 256><<<BATCH * NPTS / 16, 256, 0, stream>>>(Fin, stride, Wb[1], y2p);
                edgeconv_kernel<128, 256><<<BATCH * NPTS, 256, 0, stream>>>(
                    Fin, stride, Wb[1], g[1], bt[1], idxp, y2p, outSlice);
                break;
            case 2:
                y2_gemm_kernel<256, 512><<<BATCH * NPTS / 16, 256, 0, stream>>>(Fin, stride, Wb[2], y2p);
                edgeconv_kernel<256, 512><<<BATCH * NPTS, 256, 0, stream>>>(
                    Fin, stride, Wb[2], g[2], bt[2], idxp, y2p, outSlice);
                break;
            case 3:
                y2_gemm_kernel<512, 1024><<<BATCH * NPTS / 16, 256, 0, stream>>>(Fin, stride, Wb[3], y2p);
                edgeconv_kernel<512, 1024><<<BATCH * NPTS, 256, 0, stream>>>(
                    Fin, stride, Wb[3], g[3], bt[3], idxp, y2p, outSlice);
                break;
        }
    }

    // --- final 1x1 conv over the 1920-channel cat, f32 output (B,N,512) ---
    final_gemm_kernel<<<BATCH * (NPTS / 16), 256, 0, stream>>>(Fcat, Wlb, gl, bl, out);
}
